// GCNContextNodeClassifier_26731876451145
// MI455X (gfx1250) — matrix operand order repaired and checked by reading, not compiled, not run on hardware
//
#include <hip/hip_runtime.h>

#define NN   50000
#define EE   800000
#define HH   128
#define CTXD 128
#define BN   64
#define NC   8
#define LEPS 1e-5f

typedef __attribute__((ext_vector_type(16))) __bf16 v16bf;
typedef __attribute__((ext_vector_type(8)))  __bf16 v8bf;
typedef __attribute__((ext_vector_type(8)))  float  v8f;
typedef int v4i_vec __attribute__((vector_size(16)));   // matches builtin param "int __vector(4)"

#if __has_builtin(__builtin_amdgcn_global_load_async_to_lds_b128)
#define HAVE_ASYNC_LDS 1
#endif

// exact-typed address-space casts for the async builtin
// (AS3 pointer value = low 32 bits of generic LDS address per ISA flat->LDS rule)
#define GLB_AS1(p) ((__attribute__((address_space(1))) v4i_vec*)(unsigned long long)(p))
#define LDS_AS3(p) ((__attribute__((address_space(3))) v4i_vec*)(unsigned int)(unsigned long long)(p))

// ---------------- small utility kernels ----------------

__global__ void k_fill1(float* p, int n) {
    int i = blockIdx.x * blockDim.x + threadIdx.x;
    if (i < n) p[i] = 1.0f;
}

__global__ void k_fillbf0(__bf16* p, int n) {
    int i = blockIdx.x * blockDim.x + threadIdx.x;
    if (i < n) p[i] = (__bf16)0.0f;
}

__global__ void k_deg_accum(const int* __restrict__ col, float* __restrict__ deg, int e) {
    int i = blockIdx.x * blockDim.x + threadIdx.x;
    if (i < e) atomicAdd(&deg[col[i]], 1.0f);
}

__global__ void k_rsqrt_inplace(float* p, int n) {
    int i = blockIdx.x * blockDim.x + threadIdx.x;
    if (i < n) p[i] = rsqrtf(p[i]);
}

// dst[c*R + r] = (bf16) src[r*C + c]   (transpose + convert)
__global__ void k_transpose_bf(const float* __restrict__ src, __bf16* __restrict__ dst,
                               int R, int C) {
    int i = blockIdx.x * blockDim.x + threadIdx.x;
    if (i < R * C) {
        int r = i / C, c = i % C;
        dst[(size_t)c * R + r] = (__bf16)src[(size_t)r * C + c];
    }
}

__global__ void k_conv_bf(const float* __restrict__ src, __bf16* __restrict__ dst, int n) {
    int i = blockIdx.x * blockDim.x + threadIdx.x;
    if (i < n) dst[i] = (__bf16)src[i];
}

// ---------------- WMMA bf16 GEMM with LDS-staged A tile ----------------
// out[row, col] = sum_k A[row,k] * Wt[col,k]  (+bias[col]) (relu?)
// A: bf16 [NN x K] row-stride lda;  Wt: bf16 [Ncols x K] row-major (pre-transposed).
// Block = blockDim.x/32 waves; wave w -> col tile (blockIdx.y*(waves)+w)*16.
// The 16xK A tile is shared by all waves: staged once into LDS (async if available).
// Optional fused epilogues:
//   outF: f32 store (stride ldf);  outB: bf16 store (stride ldb_)
//   aggF: agg[row*HH+col] = bias2[col] + v * dinv[row]^2   (GCN bias + self-loop)
// ncolv: only columns < ncolv are stored (for the zero-padded 8-col head GEMM).
__global__ void k_gemm_bf16(const __bf16* __restrict__ A, int lda,
                            const __bf16* __restrict__ Wt,
                            const float* __restrict__ bias,
                            float* __restrict__ outF, int ldf,
                            __bf16* __restrict__ outB, int ldb_,
                            float* __restrict__ aggF,
                            const float* __restrict__ dinv,
                            const float* __restrict__ bias2,
                            int K, int relu, int ncolv) {
    __shared__ __bf16 As[16 * 256];   // up to K=256, 8KB

    const int lane = threadIdx.x & 31;
    const int wave = threadIdx.x >> 5;
    const int half = lane >> 4;        // 0: lanes 0-15, 1: lanes 16-31
    const int l15  = lane & 15;
    const int row0 = blockIdx.x * 16;
    const int col0 = (blockIdx.y * (blockDim.x >> 5) + wave) * 16;

    // ---- stage A tile (16 x K bf16) into LDS, 16B chunks ----
    {
        const int cpr   = K >> 3;       // 8-element chunks per row
        const int total = 16 * cpr;
        for (int ch = threadIdx.x; ch < total; ch += blockDim.x) {
            const int r  = ch / cpr;
            const int kc = (ch - r * cpr) << 3;
            const __bf16* gp = A + (size_t)(row0 + r) * lda + kc;
            __bf16* lp = &As[r * K + kc];
#ifdef HAVE_ASYNC_LDS
            __builtin_amdgcn_global_load_async_to_lds_b128(GLB_AS1(gp), LDS_AS3(lp), 0, 0);
#else
            *reinterpret_cast<v8bf*>(lp) = *reinterpret_cast<const v8bf*>(gp);
#endif
        }
#ifdef HAVE_ASYNC_LDS
#if __has_builtin(__builtin_amdgcn_s_wait_asynccnt)
        __builtin_amdgcn_s_wait_asynccnt(0);
#else
        asm volatile("s_wait_asynccnt 0" ::: "memory");
#endif
#endif
        __syncthreads();
    }

    const __bf16* asrow = &As[l15 * K];
    const __bf16* brow  = Wt + (size_t)(col0 + l15) * K;

    v8f acc = {};
    for (int kk = 0; kk < K; kk += 32) {
        // A fragment (from LDS): halves 0-7 -> K = kk + half*8 + i ; halves 8-15 -> +16
        const int ca = kk + half * 8;
        v8bf a0 = *reinterpret_cast<const v8bf*>(asrow + ca);
        v8bf a1 = *reinterpret_cast<const v8bf*>(asrow + ca + 16);
        // B fragment (from global): halves 0-7 -> K = kk + half*8 + i ; halves 8-15 -> +16
        const int cb = kk + half * 8;
        v8bf b0 = *reinterpret_cast<const v8bf*>(brow + cb);
        v8bf b1 = *reinterpret_cast<const v8bf*>(brow + cb + 16);
        v16bf va, vb;
#pragma unroll
        for (int i = 0; i < 8; ++i) {
            va[i] = a0[i]; va[i + 8] = a1[i];
            vb[i] = b0[i]; vb[i + 8] = b1[i];
        }
        acc = __builtin_amdgcn_wmma_f32_16x16x32_bf16(
            false, va, false, vb, (short)0, acc, false, false);
    }

    const int ocol = col0 + l15;
    if (ocol < ncolv) {
        const float bv = bias ? bias[ocol] : 0.0f;
#pragma unroll
        for (int r = 0; r < 8; ++r) {
            const int orow = row0 + r + half * 8;   // C/D layout: VGPR r -> M=r (+8 hi lanes)
            float v = acc[r] + bv;
            if (relu) v = v > 0.0f ? v : 0.0f;
            if (outF) outF[(size_t)orow * ldf + ocol] = v;
            if (outB) outB[(size_t)orow * ldb_ + ocol] = (__bf16)v;
            if (aggF) {
                const float d = dinv[orow];
                aggF[(size_t)orow * HH + ocol] = bias2[ocol] + v * d * d;
            }
        }
    }
}

// ---------------- graph aggregation ----------------

// one thread per (edge, 4-feature chunk): 32 chunk-threads per edge
__global__ void k_agg_edges(const int* __restrict__ rowi, const int* __restrict__ coli,
                            const float* __restrict__ dinv, const float* __restrict__ hw,
                            float* __restrict__ agg) {
    long long t = (long long)blockIdx.x * blockDim.x + threadIdx.x;
    int e = (int)(t >> 5);
    int c = ((int)t & 31) << 2;
    if (e < EE) {
        int r = rowi[e], cl = coli[e];
        float w = dinv[r] * dinv[cl];
        const float4 v = *reinterpret_cast<const float4*>(hw + (size_t)r * HH + c);
        float* dst = agg + (size_t)cl * HH + c;
        atomicAdd(dst + 0, v.x * w);
        atomicAdd(dst + 1, v.y * w);
        atomicAdd(dst + 2, v.z * w);
        atomicAdd(dst + 3, v.w * w);
    }
}

// wave-per-node LayerNorm + ReLU, writes bf16 (row stride ldo)
__global__ void k_ln_relu(const float* __restrict__ agg,
                          const float* __restrict__ g, const float* __restrict__ b,
                          __bf16* __restrict__ out, int ldo) {
    const int lane = threadIdx.x & 31;
    const int wave = threadIdx.x >> 5;
    const int node = blockIdx.x * 8 + wave;
    if (node >= NN) return;
    const float* rowp = agg + (size_t)node * HH;
    float v[4];
    float s = 0.0f;
#pragma unroll
    for (int i = 0; i < 4; ++i) { v[i] = rowp[lane + 32 * i]; s += v[i]; }
#pragma unroll
    for (int o = 16; o > 0; o >>= 1) s += __shfl_xor(s, o, 32);
    const float mu = s * (1.0f / 128.0f);
    float q = 0.0f;
#pragma unroll
    for (int i = 0; i < 4; ++i) { float d = v[i] - mu; q += d * d; }
#pragma unroll
    for (int o = 16; o > 0; o >>= 1) q += __shfl_xor(q, o, 32);
    const float inv = rsqrtf(q * (1.0f / 128.0f) + LEPS);
#pragma unroll
    for (int i = 0; i < 4; ++i) {
        int f = lane + 32 * i;
        float y = (v[i] - mu) * inv * g[f] + b[f];
        y = y > 0.0f ? y : 0.0f;
        out[(size_t)node * ldo + f] = (__bf16)y;
    }
}

// ---------------- host launcher ----------------

extern "C" void kernel_launch(void* const* d_in, const int* in_sizes, int n_in,
                              void* d_out, int out_size, void* d_ws, size_t ws_size,
                              hipStream_t stream) {
    const float* x        = (const float*)d_in[0];
    const int*   eidx     = (const int*)d_in[1];   // [2][E]
    const float* ctx      = (const float*)d_in[2];
    const float* gcn_W    = (const float*)d_in[3]; // [3][128][128]
    const float* gcn_b    = (const float*)d_in[4]; // [3][128]
    const float* ln_g     = (const float*)d_in[5];
    const float* ln_b     = (const float*)d_in[6];
    const float* ctx_W1   = (const float*)d_in[7]; // [128][64]
    const float* ctx_b1   = (const float*)d_in[8];
    const float* ctx_W2   = (const float*)d_in[9]; // [64][128]
    const float* ctx_b2   = (const float*)d_in[10];
    const float* head_W1  = (const float*)d_in[11]; // [256][128]
    const float* head_b1  = (const float*)d_in[12];
    const float* head_W2  = (const float*)d_in[13]; // [128][8]
    const float* head_b2  = (const float*)d_in[14];
    float* out = (float*)d_out;

    // workspace carve-up (256B aligned)
    char* ws = (char*)d_ws;
    size_t off = 0;
    auto carve = [&](size_t bytes) -> char* {
        char* p = ws + off;
        off = (off + bytes + 255) & ~(size_t)255;
        return p;
    };
    float*  deg     = (float*)carve((size_t)NN * 4);                 // -> dinv in place
    __bf16* wtb     = (__bf16*)carve((size_t)3 * 128 * 128 * 2);     // gcn W^T bf16
    __bf16* cW1t    = (__bf16*)carve((size_t)BN * CTXD * 2);         // [64][128]
    __bf16* cW2t    = (__bf16*)carve((size_t)HH * BN * 2);           // [128][64]
    __bf16* hW1t    = (__bf16*)carve((size_t)HH * 256 * 2);          // [128][256]
    __bf16* w2tpad  = (__bf16*)carve((size_t)16 * HH * 2);           // [16][128], rows 8..15 zero
    __bf16* h_bf    = (__bf16*)carve((size_t)NN * HH * 2);
    float*  hw      = (float*)carve((size_t)NN * HH * 4);
    float*  agg     = (float*)carve((size_t)NN * HH * 4);
    __bf16* t1      = (__bf16*)carve((size_t)NN * BN * 2);
    __bf16* fused   = (__bf16*)carve((size_t)NN * 256 * 2);
    __bf16* z_bf    = (__bf16*)carve((size_t)NN * HH * 2);           // also reused as ctx bf16
    (void)ws_size; (void)n_in; (void)in_sizes; (void)out_size;

    const int* erow = eidx;        // source (gather)
    const int* ecol = eidx + EE;   // target (scatter)

    const dim3 B256(256);

    // 1) degrees (self-loop => start at 1) -> dinv
    k_fill1<<<(NN + 255) / 256, B256, 0, stream>>>(deg, NN);
    k_deg_accum<<<(EE + 255) / 256, B256, 0, stream>>>(ecol, deg, EE);
    k_rsqrt_inplace<<<(NN + 255) / 256, B256, 0, stream>>>(deg, NN);
    float* dinv = deg;

    // 2) weight transpose + bf16 convert
    for (int l = 0; l < 3; ++l)
        k_transpose_bf<<<(128 * 128 + 255) / 256, B256, 0, stream>>>(
            gcn_W + (size_t)l * 128 * 128, wtb + (size_t)l * 128 * 128, 128, 128);
    k_transpose_bf<<<(CTXD * BN + 255) / 256, B256, 0, stream>>>(ctx_W1, cW1t, CTXD, BN);
    k_transpose_bf<<<(BN * HH + 255) / 256, B256, 0, stream>>>(ctx_W2, cW2t, BN, HH);
    k_transpose_bf<<<(256 * HH + 255) / 256, B256, 0, stream>>>(head_W1, hW1t, 256, HH);
    k_fillbf0<<<(16 * HH + 255) / 256, B256, 0, stream>>>(w2tpad, 16 * HH);
    k_transpose_bf<<<(HH * NC + 255) / 256, B256, 0, stream>>>(head_W2, w2tpad, HH, NC);

    // 3) bf16 inputs
    k_conv_bf<<<(NN * HH + 255) / 256, B256, 0, stream>>>(x, h_bf, NN * HH);
    k_conv_bf<<<(NN * CTXD + 255) / 256, B256, 0, stream>>>(ctx, z_bf, NN * CTXD);

    const int RB = NN / 16;  // 3125 row tiles

    // 4) context MLP: relu(ctx@W1+b1)@W2+b2 -> fused[:,128:256]
    k_gemm_bf16<<<dim3(RB, 1), dim3(128), 0, stream>>>(
        z_bf, CTXD, cW1t, ctx_b1, nullptr, 0, t1, BN,
        nullptr, nullptr, nullptr, CTXD, 1, BN);
    k_gemm_bf16<<<dim3(RB, 2), dim3(128), 0, stream>>>(
        t1, BN, cW2t, ctx_b2, nullptr, 0, fused + HH, 256,
        nullptr, nullptr, nullptr, BN, 0, HH);

    // 5) GCN layers: GEMM (fused agg init) -> edge scatter -> LN+ReLU
    for (int l = 0; l < 3; ++l) {
        k_gemm_bf16<<<dim3(RB, 2), dim3(128), 0, stream>>>(
            h_bf, HH, wtb + (size_t)l * 128 * 128, nullptr, hw, HH, nullptr, 0,
            agg, dinv, gcn_b + l * HH, HH, 0, HH);
        long long ework = (long long)EE * 32;
        k_agg_edges<<<(unsigned)((ework + 255) / 256), B256, 0, stream>>>(erow, ecol, dinv, hw, agg);
        __bf16* lnout = (l == 2) ? fused : h_bf;
        int     ldo   = (l == 2) ? 256 : HH;
        k_ln_relu<<<(NN + 7) / 8, B256, 0, stream>>>(agg, ln_g + l * HH, ln_b + l * HH, lnout, ldo);
    }

    // 6) head: relu(fused@W1+b1) -> z ; z@W2pad+b2 -> logits (WMMA, 8 valid cols)
    k_gemm_bf16<<<dim3(RB, 2), dim3(128), 0, stream>>>(
        fused, 256, hW1t, head_b1, nullptr, 0, z_bf, HH,
        nullptr, nullptr, nullptr, 256, 1, HH);
    k_gemm_bf16<<<dim3(RB, 1), dim3(32), 0, stream>>>(
        z_bf, HH, w2tpad, head_b2, out, NC, nullptr, 0,
        nullptr, nullptr, nullptr, HH, 0, NC);
}
